// Decoder_6090263626286
// MI455X (gfx1250) — matrix operand order repaired and checked by reading, not compiled, not run on hardware
//
#include <hip/hip_runtime.h>
#include <hip/hip_bf16.h>

// ---------------------------------------------------------------------------
// Sizes from the reference
// ---------------------------------------------------------------------------
#define VOCAB 32000
#define DE    512       // D_EMBED
#define DH    512       // D_HID
#define TT    64        // T
#define BB    32        // B
#define SS    256       // S
#define G4    (4*DH)    // 2048 gate width
#define TB    (TT*BB)   // 2048 rows
#define LDA_HW 1032     // padded row stride (bf16) for s_hw: 2064B -> bank rot 4
#define LDA_SA 520      // padded row stride (bf16) for staged A tile

typedef __bf16 bf16;
typedef __attribute__((ext_vector_type(16))) __bf16 v16bf;
typedef __attribute__((ext_vector_type(8)))  float  v8f;

// ---------------------------------------------------------------------------
// gfx1250 async global->LDS path (ASYNCcnt), guarded so compile stays safe.
// Builtin signature (from hipcc diagnostic): first arg is AS1 pointer to
// 'int __vector(4)'; mirror that for the LDS side (AS3).
// ---------------------------------------------------------------------------
#if defined(__has_builtin)
#if __has_builtin(__builtin_amdgcn_global_load_async_to_lds_b128) && \
    __has_builtin(__builtin_amdgcn_s_wait_asynccnt)
#define HAVE_ASYNC_LDS 1
#endif
#endif
#ifndef HAVE_ASYNC_LDS
#define HAVE_ASYNC_LDS 0
#endif

#if HAVE_ASYNC_LDS
typedef int v4i_t __attribute__((__vector_size__(4 * sizeof(int))));
__device__ __forceinline__ void async_copy_b128(const void* g, void* l) {
  __builtin_amdgcn_global_load_async_to_lds_b128(
      (__attribute__((address_space(1))) v4i_t*)(v4i_t*)(void*)g,
      (__attribute__((address_space(3))) v4i_t*)(v4i_t*)l, 0, 0);
}
#endif

// ---------------------------------------------------------------------------
// WMMA helpers (CDNA5 wave32 layouts, CDNA5 ISA §7.12.2)
// ---------------------------------------------------------------------------
__device__ __forceinline__ v8f wmma_bf16(v16bf a, v16bf b, v8f c) {
  return __builtin_amdgcn_wmma_f32_16x16x32_bf16(
      /*neg_a=*/false, a, /*neg_b=*/false, b,
      /*c_mod=*/(short)0, c, /*reuse_a=*/false, /*reuse_b=*/false);
}

// A fragment 16x32 bf16 from row-major A[m][k] at (row0,k0); lda in elements.
// Lane L holds row m=L&15; lanes 0-15: K={0..15}; lanes 16-31: K={16..31}.
__device__ __forceinline__ v16bf load_a(const bf16* __restrict__ A, int lda, int lane) {
  const bf16* p = A + (size_t)(lane & 15) * lda + 16 * (lane >> 4);
  v16bf r;
#pragma unroll
  for (int j = 0; j < 8; ++j) r[j] = p[j];
#pragma unroll
  for (int j = 0; j < 8; ++j) r[j + 8] = p[8 + j];
  return r;
}

// B fragment 32x16 bf16 where B[k][n] = W[n][k] (W row-major [N][K]) at (n0,k0).
// Lane L holds column n=L&15; K = 16*(L>>4)+j, contiguous -> b128+b128.
__device__ __forceinline__ v16bf load_bT(const bf16* __restrict__ W, int ldw, int lane) {
  const bf16* p = W + (size_t)(lane & 15) * ldw + 16 * (lane >> 4);
  v16bf r;
#pragma unroll
  for (int j = 0; j < 16; ++j) r[j] = p[j];
  return r;
}

// ---------------------------------------------------------------------------
// Small prep kernels
// ---------------------------------------------------------------------------
__global__ void k_f2bf(const float* __restrict__ x, bf16* __restrict__ y, size_t n) {
  size_t i = (size_t)blockIdx.x * blockDim.x + threadIdx.x;
  size_t st = (size_t)gridDim.x * blockDim.x;
  for (; i < n; i += st) y[i] = (bf16)x[i];
}

__global__ void k_bsum(const float* __restrict__ a, const float* __restrict__ b,
                       float* __restrict__ o) {
  int i = blockIdx.x * blockDim.x + threadIdx.x;
  if (i < G4) o[i] = a[i] + b[i];
}

__global__ void k_embed(const int* __restrict__ xseq, const float* __restrict__ emb,
                        bf16* __restrict__ E) {
  int tb = blockIdx.x;
  const float* src = emb + (size_t)xseq[tb] * DE;
  bf16* dst = E + (size_t)tb * DE;
  for (int d = threadIdx.x; d < DE; d += blockDim.x) dst[d] = (bf16)src[d];
}

// ---------------------------------------------------------------------------
// Input projection: Xg[TB][G4] = E @ W_ih^T + (b_ih + b_hh)
// block = 8 waves; wave owns one 16x16 tile. grid = (G4/128, TB/16)
// ---------------------------------------------------------------------------
__global__ __launch_bounds__(256) void k_gemm_inproj(
    const bf16* __restrict__ E, const bf16* __restrict__ W,
    const float* __restrict__ bsum, float* __restrict__ Xg) {
  int lane = threadIdx.x & 31, wave = threadIdx.x >> 5;
  int hi = lane >> 4, lo = lane & 15;
  int row0 = blockIdx.y * 16;
  int col0 = (blockIdx.x * 8 + wave) * 16;
  v8f acc;
  float bb = bsum[col0 + lo];
#pragma unroll
  for (int r = 0; r < 8; ++r) acc[r] = bb;
#pragma unroll 1
  for (int k0 = 0; k0 < DE; k0 += 32) {
    v16bf a = load_a(E + (size_t)row0 * DE + k0, DE, lane);
    v16bf b = load_bT(W + (size_t)col0 * DE + k0, DE, lane);
    acc = wmma_bf16(a, b, acc);
  }
  float* p = Xg + (size_t)row0 * G4 + col0;
#pragma unroll
  for (int r = 0; r < 8; ++r) p[(size_t)(r + 8 * hi) * G4 + lo] = acc[r];
}

// ---------------------------------------------------------------------------
// Output projection: L[TB][VOCAB] = outs_bf @ W_out^T + b_out (f32 logits).
// Memory-bound on the 262 MB logits store; minimize L2 A-traffic:
// block = 8 waves x 5 N-tiles = 640 cols; A tile (16x512) staged in LDS once
// per block via async global->LDS, each A fragment reused across 5 WMMAs.
// grid = (VOCAB/640 = 50, TB/16 = 128)
// ---------------------------------------------------------------------------
__global__ __launch_bounds__(256) void k_gemm_out(
    const bf16* __restrict__ A, const bf16* __restrict__ W,
    const float* __restrict__ bias, float* __restrict__ L) {
  __shared__ bf16 sA[16 * LDA_SA];
  const int tid = threadIdx.x, lane = tid & 31, wave = tid >> 5;
  const int hi = lane >> 4, lo = lane & 15;
  const int row0 = blockIdx.y * 16;

  // stage A tile: 16 rows x 64 chunks of 8 bf16 (b128 each)
  {
    const bf16* src = A + (size_t)row0 * DH;
#pragma unroll
    for (int i = tid; i < 16 * 64; i += 256) {
      int rr = i >> 6, cc = i & 63;
#if HAVE_ASYNC_LDS
      async_copy_b128(src + rr * DH + cc * 8, sA + rr * LDA_SA + cc * 8);
#else
      *(uint4*)(sA + rr * LDA_SA + cc * 8) = *(const uint4*)(src + rr * DH + cc * 8);
#endif
    }
#if HAVE_ASYNC_LDS
    __builtin_amdgcn_s_wait_asynccnt(0);
#endif
    __syncthreads();
  }

  const int colbase = (blockIdx.x * 40 + wave * 5) * 16;  // 5 tiles per wave
  v8f acc[5];
#pragma unroll
  for (int q = 0; q < 5; ++q) {
    float bb = bias[colbase + q * 16 + lo];
#pragma unroll
    for (int r = 0; r < 8; ++r) acc[q][r] = bb;
  }
#pragma unroll 1
  for (int k0 = 0; k0 < DH; k0 += 32) {
    v16bf a = load_a(sA + k0, LDA_SA, lane);  // LDS, conflict-free (pad)
#pragma unroll
    for (int q = 0; q < 5; ++q) {
      v16bf b = load_bT(W + (size_t)(colbase + q * 16) * DH + k0, DH, lane);
      acc[q] = wmma_bf16(a, b, acc[q]);
    }
  }
#pragma unroll
  for (int q = 0; q < 5; ++q) {
    float* p = L + (size_t)row0 * VOCAB + colbase + q * 16;
#pragma unroll
    for (int r = 0; r < 8; ++r) p[(size_t)(r + 8 * hi) * VOCAB + lo] = acc[q][r];
  }
}

// ---------------------------------------------------------------------------
// Row-wise log-softmax over logits in place. grid = TB rows, block = 256.
// ---------------------------------------------------------------------------
__global__ __launch_bounds__(256) void k_logsoftmax(float* __restrict__ L) {
  float* p = L + (size_t)blockIdx.x * VOCAB;
  int tid = threadIdx.x, lane = tid & 31, wv = tid >> 5;
  __shared__ float redm[8], reds[8];
  float m = -3.4e38f;
  for (int i = tid; i < VOCAB; i += 256) m = fmaxf(m, p[i]);
#pragma unroll
  for (int o = 16; o; o >>= 1) m = fmaxf(m, __shfl_xor(m, o, 32));
  if (lane == 0) redm[wv] = m;
  __syncthreads();
  float mm = redm[0];
#pragma unroll
  for (int i = 1; i < 8; ++i) mm = fmaxf(mm, redm[i]);
  float s = 0.f;
  for (int i = tid; i < VOCAB; i += 256) s += __expf(p[i] - mm);
#pragma unroll
  for (int o = 16; o; o >>= 1) s += __shfl_xor(s, o, 32);
  if (lane == 0) reds[wv] = s;
  __syncthreads();
  float ss = 0.f;
#pragma unroll
  for (int i = 0; i < 8; ++i) ss += reds[i];
  float lse = mm + __logf(ss);
  for (int i = tid; i < VOCAB; i += 256) p[i] = p[i] - lse;
}

// ---------------------------------------------------------------------------
// Recurrence: one workgroup (1024 threads = 32 wave32s) runs all 64 steps.
// LDS (dynamic, ~160 KB of the WGP's 320 KB), padded strides for bank spread:
//   s_hw [32][1032] bf16 : concat(h, wctx) -> WMMA A fragments
//   s_h  [32][512]  f32  : h for attention dots
//   s_sc [32][256]  f32  : scores / attn weights
// Gates GEMM: wave owns fixed M-tile (mt = wave&1); k-outer loop loads each A
// fragment once and feeds 4 accumulators (4 independent WMMA chains).
// ---------------------------------------------------------------------------
__global__ __launch_bounds__(1024) void k_recurrence(
    const float* __restrict__ h0, const float* __restrict__ c0,
    const float* __restrict__ context,          // [32][256][512] f32
    const bf16* __restrict__ Whh,               // [2048][512]
    const bf16* __restrict__ Wattn,             // [512][1024]
    const float* __restrict__ Xg,               // [2048][2048]
    float* __restrict__ gbuf,                   // [32][2048] scratch
    bf16* __restrict__ outs,                    // [2048][512]
    float* __restrict__ hf, float* __restrict__ cf) {
  extern __shared__ char smem[];
  bf16*  s_hw = (bf16*)smem;                           // 32*1032*2 = 66048 B
  float* s_h  = (float*)(smem + 66048);                // 65536 B
  float* s_sc = (float*)(smem + 66048 + 65536);        // 32768 B

  const int tid = threadIdx.x;
  const int lane = tid & 31, wave = tid >> 5;
  const int hi = lane >> 4, lo = lane & 15;
  const int mt = wave & 1;                // fixed M-tile per wave
  const int ntg = (wave >> 1) << 3;       // base N-tile: 0,8,...,120

  for (int i = tid; i < BB * DH; i += 1024) {
    int b = i >> 9, d = i & 511;
    float v = h0[i];
    s_h[i] = v;
    s_hw[b * LDA_HW + d] = (bf16)v;
  }
  float creg[16];
#pragma unroll
  for (int r = 0; r < 16; ++r) creg[r] = c0[tid + (r << 10)];
  __syncthreads();

  for (int t = 0; t < TT; ++t) {
    // ---- gates = Xg[t] + h @ W_hh^T : 2x128 tiles; 8 per wave in 2 groups of 4
#pragma unroll 1
    for (int g = 0; g < 2; ++g) {
      v8f acc[4];
      const float* xg = Xg + (size_t)(t * BB + mt * 16) * G4;
#pragma unroll
      for (int q = 0; q < 4; ++q) {
        int col0 = (ntg + g * 4 + q) << 4;
#pragma unroll
        for (int r = 0; r < 8; ++r) acc[q][r] = xg[(size_t)(r + 8 * hi) * G4 + col0 + lo];
      }
#pragma unroll 1
      for (int k0 = 0; k0 < DH; k0 += 32) {
        v16bf a = load_a(s_hw + (mt * 16) * LDA_HW + k0, LDA_HW, lane);  // 1 ds frag
#pragma unroll
        for (int q = 0; q < 4; ++q) {                                    // 4 WMMAs
          v16bf b = load_bT(Whh + (size_t)((ntg + g * 4 + q) << 4) * DH + k0, DH, lane);
          acc[q] = wmma_bf16(a, b, acc[q]);
        }
      }
#pragma unroll
      for (int q = 0; q < 4; ++q) {
        float* gp = gbuf + (size_t)(mt * 16) * G4 + ((ntg + g * 4 + q) << 4);
#pragma unroll
        for (int r = 0; r < 8; ++r) gp[(size_t)(r + 8 * hi) * G4 + lo] = acc[q][r];
      }
    }
    __syncthreads();

    // ---- LSTM elementwise: 16 cells/thread, c persistent in VGPRs ----
#pragma unroll
    for (int r = 0; r < 16; ++r) {
      int idx = tid + (r << 10);              // b*512 + j
      int b = idx >> 9, j = idx & 511;
      const float* g = gbuf + (size_t)b * G4;
      float i_ = 1.f / (1.f + __expf(-g[j]));
      float f_ = 1.f / (1.f + __expf(-g[j + 512]));
      float gg = tanhf(g[j + 1024]);
      float o_ = 1.f / (1.f + __expf(-g[j + 1536]));
      float c = f_ * creg[r] + i_ * gg;
      creg[r] = c;
      float h = o_ * tanhf(c);
      s_h[idx] = h;
      s_hw[b * LDA_HW + j] = (bf16)h;
    }
    __syncthreads();

    // ---- attention scores: (b,s) pairs, 8 per thread ----
#pragma unroll 1
    for (int r = 0; r < 8; ++r) {
      int p = tid + (r << 10);
      int b = p >> 8, s = p & 255;
      const float* ctx = context + ((size_t)b * SS + s) * DH;
      const float* hh = s_h + b * DH;
      float acc = 0.f;
      for (int d = 0; d < DH; ++d) acc = fmaf(ctx[d], hh[d], acc);
      s_sc[p] = acc;
    }
    __syncthreads();

    // ---- softmax over s: wave w handles batch row b = w (wave32) ----
    {
      float* row = s_sc + wave * SS;
      float mx = -3.4e38f;
#pragma unroll
      for (int r = 0; r < 8; ++r) mx = fmaxf(mx, row[lane + 32 * r]);
#pragma unroll
      for (int o = 16; o; o >>= 1) mx = fmaxf(mx, __shfl_xor(mx, o, 32));
      float ev[8], sum = 0.f;
#pragma unroll
      for (int r = 0; r < 8; ++r) { ev[r] = __expf(row[lane + 32 * r] - mx); sum += ev[r]; }
#pragma unroll
      for (int o = 16; o; o >>= 1) sum += __shfl_xor(sum, o, 32);
      float inv = 1.f / sum;
#pragma unroll
      for (int r = 0; r < 8; ++r) row[lane + 32 * r] = ev[r] * inv;
    }
    __syncthreads();

    // ---- weighted context -> bf16 concat half ----
#pragma unroll 1
    for (int r = 0; r < 16; ++r) {
      int idx = tid + (r << 10);
      int b = idx >> 9, d = idx & 511;
      const float* ctx = context + (size_t)b * SS * DH + d;
      const float* at = s_sc + b * SS;
      float acc = 0.f;
      for (int s = 0; s < SS; ++s) acc = fmaf(at[s], ctx[(size_t)s * DH], acc);
      s_hw[b * LDA_HW + DH + d] = (bf16)acc;
    }
    __syncthreads();

    // ---- a = tanh(concat @ W_attn^T): 2x32 tiles, 2 per wave, A reused ----
    {
      const int nt0 = wave >> 1, nt1 = nt0 + 16;   // mt = wave&1 fixed
      v8f a0 = {}, a1 = {};
#pragma unroll 1
      for (int k0 = 0; k0 < 2 * DH; k0 += 32) {
        v16bf a = load_a(s_hw + (mt * 16) * LDA_HW + k0, LDA_HW, lane);
        v16bf b0 = load_bT(Wattn + (size_t)(nt0 * 16) * (2 * DH) + k0, 2 * DH, lane);
        a0 = wmma_bf16(a, b0, a0);
        v16bf b1 = load_bT(Wattn + (size_t)(nt1 * 16) * (2 * DH) + k0, 2 * DH, lane);
        a1 = wmma_bf16(a, b1, a1);
      }
      bf16* op = outs + (size_t)(t * BB + mt * 16) * DH;
#pragma unroll
      for (int r = 0; r < 8; ++r) {
        op[(size_t)(r + 8 * hi) * DH + nt0 * 16 + lo] = (bf16)tanhf(a0[r]);
        op[(size_t)(r + 8 * hi) * DH + nt1 * 16 + lo] = (bf16)tanhf(a1[r]);
      }
    }
    __syncthreads();
  }

  // final h, c
  for (int i = tid; i < BB * DH; i += 1024) hf[i] = s_h[i];
#pragma unroll
  for (int r = 0; r < 16; ++r) cf[tid + (r << 10)] = creg[r];
}

// ---------------------------------------------------------------------------
// Host orchestration
// ---------------------------------------------------------------------------
extern "C" void kernel_launch(void* const* d_in, const int* in_sizes, int n_in,
                              void* d_out, int out_size, void* d_ws, size_t ws_size,
                              hipStream_t stream) {
  const int*   x_seq  = (const int*)  d_in[0];
  const float* h0     = (const float*)d_in[1];
  const float* c0     = (const float*)d_in[2];
  const float* ctx    = (const float*)d_in[3];
  const float* emb    = (const float*)d_in[4];
  const float* W_ih   = (const float*)d_in[5];
  const float* W_hh   = (const float*)d_in[6];
  const float* b_ih   = (const float*)d_in[7];
  const float* b_hh   = (const float*)d_in[8];
  const float* W_attn = (const float*)d_in[9];
  const float* W_out  = (const float*)d_in[10];
  const float* b_out  = (const float*)d_in[11];

  float* out = (float*)d_out;
  float* logp = out;                          // [TB][VOCAB]
  float* hf = out + (size_t)TB * VOCAB;       // [BB*DH]
  float* cf = hf + BB * DH;                   // [BB*DH]

  char* ws = (char*)d_ws;
  bf16*  Ebf     = (bf16*)(ws);                                  // 2 MB
  bf16*  Wih_bf  = (bf16*)(ws + (2u << 20));                     // 2 MB
  bf16*  Whh_bf  = (bf16*)(ws + (4u << 20));                     // 2 MB
  bf16*  Wat_bf  = (bf16*)(ws + (6u << 20));                     // 1 MB
  bf16*  Wout_bf = (bf16*)(ws + (7u << 20));                     // 32 MB
  float* bsum    = (float*)(ws + (39u << 20));                   // 8 KB
  float* Xg      = (float*)(ws + (40u << 20));                   // 16 MB
  float* gbuf    = (float*)(ws + (56u << 20));                   // 256 KB
  bf16*  outs_bf = (bf16*)(ws + (57u << 20));                    // 2 MB (59 MB total)

  k_embed<<<TB, 256, 0, stream>>>(x_seq, emb, Ebf);
  k_f2bf<<<1024, 256, 0, stream>>>(W_ih, Wih_bf, (size_t)G4 * DE);
  k_f2bf<<<1024, 256, 0, stream>>>(W_hh, Whh_bf, (size_t)G4 * DH);
  k_f2bf<<<1024, 256, 0, stream>>>(W_attn, Wat_bf, (size_t)DH * 2 * DH);
  k_f2bf<<<4096, 256, 0, stream>>>(W_out, Wout_bf, (size_t)VOCAB * DH);
  k_bsum<<<(G4 + 255) / 256, 256, 0, stream>>>(b_ih, b_hh, bsum);

  k_gemm_inproj<<<dim3(G4 / 128, TB / 16), 256, 0, stream>>>(Ebf, Wih_bf, bsum, Xg);

  (void)hipFuncSetAttribute((const void*)k_recurrence,
                            hipFuncAttributeMaxDynamicSharedMemorySize, 66048 + 65536 + 32768);
  k_recurrence<<<1, 1024, 66048 + 65536 + 32768, stream>>>(h0, c0, ctx, Whh_bf, Wat_bf, Xg,
                                                           gbuf, outs_bf, hf, cf);

  k_gemm_out<<<dim3(VOCAB / 640, TB / 16), 256, 0, stream>>>(outs_bf, Wout_bf, b_out, logp);
  k_logsoftmax<<<TB, 256, 0, stream>>>(logp);
}
